// MambaMoELayer_38998303047961
// MI455X (gfx1250) — matrix operand order repaired and checked by reading, not compiled, not run on hardware
//
#include <hip/hip_runtime.h>
#include <hip/hip_bf16.h>

// ---------------- types ----------------
typedef __bf16 bf16_t;
typedef bf16_t v16bf __attribute__((ext_vector_type(16)));
typedef bf16_t v8bf  __attribute__((ext_vector_type(8)));
typedef float  v8f   __attribute__((ext_vector_type(8)));

union Frag { v16bf v; v8bf h[2]; };

// ---------------- problem constants ----------------
#define B_    4
#define L_    1024
#define DM    512          // D_MODEL
#define DI    1024         // D_INNER
#define DST   16           // D_STATE
#define DTR   32           // DT_RANK
#define XZLD  2048         // 2*D_INNER
#define TOK   (B_*L_)      // 4096 tokens

// =====================================================================
// WMMA GEMM:  C[M,N] = A[M,K] * B[N,K]^T  (A,B bf16 row-major, f32 acc)
// Wave tile: 32 x (NI*16)  (2 x NI accumulators of v_wmma_f32_16x16x32_bf16)
// Block tile: (WM*32) x (WN*NI*16), WM*WN waves.
// EPI: 0 = store f32
//      1 = +bias, softplus, store f32
//      2 = +bias, relu, store bf16
//      3 = 0.5*(v+bias) + skip  -> store f32   (MoE expert 0)
//      4 = accumulate 0.5*(v+bias) into f32    (MoE expert 1)
//      5 = store f32 AND bf16
// =====================================================================
template<int WM, int WN, int NI, int EPI>
__global__ __launch_bounds__(WM*WN*32)
void gemm_bf16_wmma(const bf16_t* __restrict__ A, const bf16_t* __restrict__ B,
                    const float* __restrict__ bias, const float* __restrict__ skip,
                    float* __restrict__ Cf, bf16_t* __restrict__ Cb,
                    int M, int N, int K, int lda, int ldb, int ldc)
{
    const int lane = threadIdx.x & 31;
    const int wid  = threadIdx.x >> 5;
    const int wm   = wid / WN, wn = wid % WN;
    const int l15  = lane & 15, lhi = lane >> 4;

    const int row0 = blockIdx.y * (WM*32)    + wm*32;
    const int col0 = blockIdx.x * (WN*NI*16) + wn*(NI*16);

    v8f acc[2][NI] = {};

    const bf16_t* Ar[2];
    Ar[0] = A + (size_t)(row0 +      l15) * lda;
    Ar[1] = A + (size_t)(row0 + 16 + l15) * lda;
    const bf16_t* Br[NI];
    #pragma unroll
    for (int ni = 0; ni < NI; ++ni)
        Br[ni] = B + (size_t)(col0 + ni*16 + l15) * ldb;

    for (int k0 = 0; k0 < K; k0 += 32) {
        // A fragment: lanes 0-15 hold K = k..k+7 (elems 0-7) and K = k+16..k+23
        // (elems 8-15); lanes 16-31 hold K = k+8..k+15 / k+24..k+31.
        const int ka = k0 + lhi*8;
        // B fragment: same K layout as A (lanes 0-15: k..k+7 / k+16..k+23; lanes 16-31: k+8..k+15 / k+24..k+31).
        const int kb = k0 + lhi*8;

        Frag a[2], b[NI];
        #pragma unroll
        for (int mi = 0; mi < 2; ++mi) {
            a[mi].h[0] = *(const v8bf*)(Ar[mi] + ka);
            a[mi].h[1] = *(const v8bf*)(Ar[mi] + ka + 16);
        }
        #pragma unroll
        for (int ni = 0; ni < NI; ++ni) {
            b[ni].h[0] = *(const v8bf*)(Br[ni] + kb);
            b[ni].h[1] = *(const v8bf*)(Br[ni] + kb + 16);
        }

        if (k0 + 32 < K) {                        // gfx1250 global_prefetch_b8
            __builtin_prefetch(Ar[0] + ka + 32, 0, 3);
            __builtin_prefetch(Ar[1] + ka + 32, 0, 3);
            #pragma unroll
            for (int ni = 0; ni < NI; ++ni)
                __builtin_prefetch(Br[ni] + kb + 32, 0, 3);
        }

        #pragma unroll
        for (int mi = 0; mi < 2; ++mi)
        #pragma unroll
        for (int ni = 0; ni < NI; ++ni)
            acc[mi][ni] = __builtin_amdgcn_wmma_f32_16x16x32_bf16(
                false, a[mi].v, false, b[ni].v, (short)0, acc[mi][ni], false, false);
    }

    // D layout: VGPR r -> row (r + 8*lhi), lane l15 -> col
    #pragma unroll
    for (int mi = 0; mi < 2; ++mi)
    #pragma unroll
    for (int ni = 0; ni < NI; ++ni) {
        const int n = col0 + ni*16 + l15;
        #pragma unroll
        for (int r = 0; r < 8; ++r) {
            const int m = row0 + mi*16 + lhi*8 + r;
            const size_t o = (size_t)m * ldc + n;
            float v = acc[mi][ni][r];
            if constexpr (EPI == 0) {
                Cf[o] = v;
            } else if constexpr (EPI == 1) {
                float t = v + bias[n];
                Cf[o] = (t > 20.f) ? t : log1pf(__expf(t));   // softplus
            } else if constexpr (EPI == 2) {
                float t = v + bias[n];
                Cb[o] = (bf16_t)fmaxf(t, 0.f);
            } else if constexpr (EPI == 3) {
                Cf[o] = 0.5f * (v + bias[n]) + skip[o];
            } else if constexpr (EPI == 4) {
                Cf[o] += 0.5f * (v + bias[n]);
            } else if constexpr (EPI == 5) {
                Cf[o] = v;
                Cb[o] = (bf16_t)v;
            }
        }
    }
}

// =====================================================================
// LayerNorm over last dim (D=512). One token per 256-thread block,
// 2 elements/thread. Optional second input (residual add), optional
// f32 / bf16 outputs.
// =====================================================================
__global__ __launch_bounds__(256)
void k_layernorm(const float* __restrict__ in1, const float* __restrict__ in2,
                 float* __restrict__ outF, bf16_t* __restrict__ outB)
{
    __shared__ float red[256];
    const int tid = threadIdx.x;
    const size_t base = (size_t)blockIdx.x * DM;

    float v0 = in1[base + tid];
    float v1 = in1[base + tid + 256];
    if (in2) { v0 += in2[base + tid]; v1 += in2[base + tid + 256]; }

    red[tid] = v0 + v1;
    __syncthreads();
    for (int s = 128; s > 0; s >>= 1) { if (tid < s) red[tid] += red[tid + s]; __syncthreads(); }
    const float mu = red[0] * (1.f / DM);
    __syncthreads();

    const float d0 = v0 - mu, d1 = v1 - mu;
    red[tid] = d0*d0 + d1*d1;
    __syncthreads();
    for (int s = 128; s > 0; s >>= 1) { if (tid < s) red[tid] += red[tid + s]; __syncthreads(); }
    const float rstd = rsqrtf(red[0] * (1.f / DM) + 1e-5f);

    const float o0 = d0 * rstd, o1 = d1 * rstd;
    if (outF) { outF[base + tid] = o0; outF[base + tid + 256] = o1; }
    if (outB) { outB[base + tid] = (bf16_t)o0; outB[base + tid + 256] = (bf16_t)o1; }
}

// =====================================================================
// Depthwise causal conv (k=4) along L + bias + SiLU. xc = xz[:, :DI].
// =====================================================================
__global__ __launch_bounds__(256)
void k_conv_silu(const float* __restrict__ xz, const float* __restrict__ cw,
                 const float* __restrict__ cb, float* __restrict__ xsF,
                 bf16_t* __restrict__ xsB)
{
    const int idx = blockIdx.x * 256 + threadIdx.x;   // over B*L*DI
    if (idx >= B_ * L_ * DI) return;
    const int d = idx % DI;
    const int t = (idx / DI) % L_;
    const int b = idx / (DI * L_);

    float acc = cb[d];
    const float* base = xz + (size_t)b * L_ * XZLD + d;
    #pragma unroll
    for (int j = 0; j < 4; ++j) {
        const int l = t - 3 + j;
        if (l >= 0) acc += base[(size_t)l * XZLD] * cw[d*4 + j];
    }
    const float s = acc / (1.f + __expf(-acc));       // silu
    xsF[idx] = s;
    xsB[idx] = (bf16_t)s;
}

// =====================================================================
// f32 -> bf16 conversion (weights)
// =====================================================================
__global__ __launch_bounds__(256)
void k_cvt_bf16(const float* __restrict__ in, bf16_t* __restrict__ out, int n)
{
    const int i = blockIdx.x * 256 + threadIdx.x;
    if (i < n) out[i] = (bf16_t)in[i];
}

// =====================================================================
// Selective scan. 16 lanes per channel (lane n = state index). Each
// block = 256 threads = 16 channels. Cross-lane reduce via shfl_xor
// (masks 1,2,4,8 stay inside the 16-lane group; wave32-safe).
// Fused epilogue: y = (sum_n h*C + xs*D) * silu(z)  -> bf16.
// =====================================================================
__global__ __launch_bounds__(256)
void k_scan(const float* __restrict__ delta, const float* __restrict__ xsF,
            const float* __restrict__ dbc, const float* __restrict__ xz,
            const float* __restrict__ A_log, const float* __restrict__ D_skip,
            bf16_t* __restrict__ ygate)
{
    const int tid  = threadIdx.x;
    const int n    = tid & 15;
    const int g    = tid >> 4;
    const int chan = blockIdx.x * 16 + g;      // 0 .. B*DI-1
    const int b    = chan / DI;
    const int d    = chan % DI;

    const float a  = -__expf(A_log[d * DST + n]);
    const float Dd = D_skip[d];
    float h = 0.f;

    for (int l = 0; l < L_; ++l) {
        const size_t row = (size_t)b * L_ + l;
        const float dlt = delta[row * DI + d];
        const float xsv = xsF[row * DI + d];
        const float Bv  = dbc[row * 64 + DTR + n];
        const float Cv  = dbc[row * 64 + DTR + DST + n];

        h = __expf(dlt * a) * h + dlt * xsv * Bv;
        float p = h * Cv;
        p += __shfl_xor(p, 1, 32);
        p += __shfl_xor(p, 2, 32);
        p += __shfl_xor(p, 4, 32);
        p += __shfl_xor(p, 8, 32);

        if (n == 0) {
            const float zv = xz[row * XZLD + DI + d];
            const float sz = zv / (1.f + __expf(-zv));
            ygate[row * DI + d] = (bf16_t)((p + xsv * Dd) * sz);
        }
    }
}

// =====================================================================
// Host-side pipeline
// =====================================================================
extern "C" void kernel_launch(void* const* d_in, const int* in_sizes, int n_in,
                              void* d_out, int out_size, void* d_ws, size_t ws_size,
                              hipStream_t stream)
{
    const float* x    = (const float*)d_in[0];
    const float* Wip  = (const float*)d_in[1];   // [2048,512]
    const float* cw   = (const float*)d_in[2];   // [1024,1,4]
    const float* cb   = (const float*)d_in[3];   // [1024]
    const float* Wxp  = (const float*)d_in[4];   // [64,1024]
    const float* Wdt  = (const float*)d_in[5];   // [1024,32]
    const float* bdt  = (const float*)d_in[6];   // [1024]
    const float* Alog = (const float*)d_in[7];   // [1024,16]
    const float* Dsk  = (const float*)d_in[8];   // [1024]
    const float* Wop  = (const float*)d_in[9];   // [512,1024]
    const float* W1   = (const float*)d_in[10];  // [2,1024,512]
    const float* b1   = (const float*)d_in[11];  // [2,1024]
    const float* W2   = (const float*)d_in[12];  // [2,512,1024]
    const float* b2   = (const float*)d_in[13];  // [2,512]
    float* out = (float*)d_out;                  // [4096,512] f32

    // ---- workspace carve-out (256B aligned) ----
    char* p = (char*)d_ws;
    auto alloc = [&](size_t bytes) { char* r = p; p += (bytes + 255) & ~(size_t)255; return r; };
    bf16_t* WipB = (bf16_t*)alloc((size_t)XZLD*DM*2);
    bf16_t* WxpB = (bf16_t*)alloc((size_t)64*DI*2);
    bf16_t* WdtB = (bf16_t*)alloc((size_t)DI*DTR*2);
    bf16_t* WopB = (bf16_t*)alloc((size_t)DM*DI*2);
    bf16_t* W1B  = (bf16_t*)alloc((size_t)2*DI*DM*2);
    bf16_t* W2B  = (bf16_t*)alloc((size_t)2*DM*DI*2);
    float*  ln1F = (float* )alloc((size_t)TOK*DM*4);
    bf16_t* ln1B = (bf16_t*)alloc((size_t)TOK*DM*2);
    float*  xzF  = (float* )alloc((size_t)TOK*XZLD*4);
    float*  xsF  = (float* )alloc((size_t)TOK*DI*4);
    bf16_t* xsB  = (bf16_t*)alloc((size_t)TOK*DI*2);
    float*  dbcF = (float* )alloc((size_t)TOK*64*4);
    bf16_t* dbcB = (bf16_t*)alloc((size_t)TOK*64*2);
    float*  dltF = (float* )alloc((size_t)TOK*DI*4);
    bf16_t* ygB  = (bf16_t*)alloc((size_t)TOK*DI*2);
    float*  mamF = (float* )alloc((size_t)TOK*DM*4);
    bf16_t* h3B  = (bf16_t*)alloc((size_t)TOK*DM*2);
    bf16_t* aB   = (bf16_t*)alloc((size_t)TOK*DI*2);
    (void)ws_size; (void)in_sizes; (void)n_in; (void)out_size;

    auto cvt = [&](const float* src, bf16_t* dst, int n) {
        k_cvt_bf16<<<(n + 255) / 256, 256, 0, stream>>>(src, dst, n);
    };

    // 0) weight conversions (recomputed every call; deterministic)
    cvt(Wip, WipB, XZLD*DM);
    cvt(Wxp, WxpB, 64*DI);
    cvt(Wdt, WdtB, DI*DTR);
    cvt(Wop, WopB, DM*DI);
    cvt(W1,  W1B,  2*DI*DM);
    cvt(W2,  W2B,  2*DM*DI);

    // 1) ln1 = LN(x)
    k_layernorm<<<TOK, 256, 0, stream>>>(x, nullptr, ln1F, ln1B);

    // 2) xz = ln1 @ in_proj^T   [4096,2048]   (block tile 64x256)
    gemm_bf16_wmma<2,4,4,0><<<dim3(XZLD/256, TOK/64), 256, 0, stream>>>(
        ln1B, WipB, nullptr, nullptr, xzF, nullptr, TOK, XZLD, DM, DM, DM, XZLD);

    // 3) xs = silu(causal_conv(xc) + cb)
    k_conv_silu<<<(B_*L_*DI)/256, 256, 0, stream>>>(xzF, cw, cb, xsF, xsB);

    // 4) dbc = xs @ x_proj^T    [4096,64]  (block tile 128x64; f32 + bf16 out)
    gemm_bf16_wmma<4,2,2,5><<<dim3(64/64, TOK/128), 256, 0, stream>>>(
        xsB, WxpB, nullptr, nullptr, dbcF, dbcB, TOK, 64, DI, DI, DI, 64);

    // 5) delta = softplus(dt @ dt_proj^T + bdt)   (A = dbc cols 0..31, lda=64, K=32)
    gemm_bf16_wmma<2,4,4,1><<<dim3(DI/256, TOK/64), 256, 0, stream>>>(
        dbcB, WdtB, bdt, nullptr, dltF, nullptr, TOK, DI, DTR, 64, DTR, DI);

    // 6) selective scan + gate -> ygate bf16
    k_scan<<<(B_*DI)/16, 256, 0, stream>>>(dltF, xsF, dbcF, xzF, Alog, Dsk, ygB);

    // 7) mamba_out = ygate @ out_proj^T   [4096,512]
    gemm_bf16_wmma<2,4,4,0><<<dim3(DM/256, TOK/64), 256, 0, stream>>>(
        ygB, WopB, nullptr, nullptr, mamF, nullptr, TOK, DM, DI, DI, DI, DM);

    // 8) h3 = LN(mamba_out + ln1)
    k_layernorm<<<TOK, 256, 0, stream>>>(mamF, ln1F, nullptr, h3B);

    // 9) MoE: out = 0.5*sum_e (relu(h3 W1e^T + b1e) W2e^T + b2e) + x
    for (int e = 0; e < 2; ++e) {
        gemm_bf16_wmma<2,4,4,2><<<dim3(DI/256, TOK/64), 256, 0, stream>>>(
            h3B, W1B + (size_t)e*DI*DM, b1 + e*DI, nullptr, nullptr, aB,
            TOK, DI, DM, DM, DM, DI);
        if (e == 0) {
            gemm_bf16_wmma<2,4,4,3><<<dim3(DM/256, TOK/64), 256, 0, stream>>>(
                aB, W2B + (size_t)e*DM*DI, b2 + e*DM, x, out, nullptr,
                TOK, DM, DI, DI, DI, DM);
        } else {
            gemm_bf16_wmma<2,4,4,4><<<dim3(DM/256, TOK/64), 256, 0, stream>>>(
                aB, W2B + (size_t)e*DM*DI, b2 + e*DM, nullptr, out, nullptr,
                TOK, DM, DI, DI, DI, DM);
        }
    }
}
